// ModelStage_42949672961923
// MI455X (gfx1250) — matrix operand order repaired and checked by reading, not compiled, not run on hardware
//
#include <hip/hip_runtime.h>

typedef __attribute__((ext_vector_type(4)))  float    f32x4;
typedef __attribute__((ext_vector_type(8)))  float    v8f;
typedef __attribute__((ext_vector_type(8)))  _Float16 h8;
typedef __attribute__((ext_vector_type(16))) _Float16 v16h;

#define NB   4
#define C    64
#define P    63
#define H    128
#define W    128
#define HP   132
#define WP   132

enum { MODE_REP = 0, MODE_ZERO = 1, MODE_GRAD = 2 };

// ---------------------------------------------------------------------------
// Build f16 weight tables.
// Wf[l][tap][co][ci] = f_rest[l][co][ci][kh][kw]              (forward conv A)
// Wb[l][tap][ci][co] = f_rest[l][co][ci][4-kh][4-kw]          (transpose conv A)
// ---------------------------------------------------------------------------
__global__ __launch_bounds__(256) void prep_weights_kernel(
    const float* __restrict__ fr, _Float16* __restrict__ Wf, _Float16* __restrict__ Wb) {
  const int per = 3 * 25 * 64 * 64;
  int gid = blockIdx.x * 256 + threadIdx.x;
  if (gid >= 2 * per) return;
  int which = (gid >= per);
  int q = which ? gid - per : gid;
  int inner = q & 63;            // ci for Wf, co for Wb
  int mid   = (q >> 6) & 63;     // co for Wf, ci for Wb
  int tap   = (q >> 12) % 25;
  int l     = (q >> 12) / 25;
  int kh = tap / 5, kw = tap % 5;
  if (!which) {
    int co = mid, ci = inner;
    Wf[q] = (_Float16)fr[(((l * 64 + co) * 64 + ci) * 25) + tap];
  } else {
    int ci = mid, co = inner;
    int tapf = (4 - kh) * 5 + (4 - kw);
    Wb[q] = (_Float16)fr[(((l * 64 + co) * 64 + ci) * 25) + tapf];
  }
}

// ---------------------------------------------------------------------------
// Stage 0: 1->64 conv with replication pad, x scaled by 255, +bias. NHWC out.
// ---------------------------------------------------------------------------
__global__ __launch_bounds__(256) void conv0_kernel(
    const float* __restrict__ x, const float* __restrict__ f0,
    const float* __restrict__ bias0, float* __restrict__ t0) {
  int gid = blockIdx.x * 256 + threadIdx.x;
  if (gid >= NB * H * W * 8) return;
  int c0 = (gid & 7) * 8;
  int pix = gid >> 3;
  int w = pix % W, h = (pix / W) % H, n = pix / (W * H);
  float s[8];
#pragma unroll
  for (int j = 0; j < 8; ++j) s[j] = 0.f;
#pragma unroll
  for (int kh = 0; kh < 5; ++kh) {
    int hc = min(max(h + kh - 2, 0), H - 1);
#pragma unroll
    for (int kw = 0; kw < 5; ++kw) {
      int wc = min(max(w + kw - 2, 0), W - 1);
      float xv = x[(n * H + hc) * W + wc];
      int tap = kh * 5 + kw;
#pragma unroll
      for (int j = 0; j < 8; ++j)
        s[j] = fmaf(xv, f0[(c0 + j) * 25 + tap], s[j]);
    }
  }
  float* o = t0 + (size_t)pix * 64 + c0;
#pragma unroll
  for (int j = 0; j < 8; ++j) o[j] = 255.f * s[j] + bias0[c0 + j];
}

// ---------------------------------------------------------------------------
// Fused RBF activation (+ optional gradient * g gating) + pad + f32->f16.
//   MODE_REP : forward  act, replication border
//   MODE_ZERO: forward  act, zero border (input to first transpose conv)
//   MODE_GRAD: g * act'(t), zero border
// Output NHWC padded [NB][132][132][64] f16.
// ---------------------------------------------------------------------------
template <int MODE>
__global__ __launch_bounds__(256) void act_pad_kernel(
    const float* __restrict__ t, const float* __restrict__ aw,
    const float* __restrict__ g, _Float16* __restrict__ pad) {
  __shared__ float sw[C * P];
  for (int j = threadIdx.x; j < C * P; j += 256) sw[j] = aw[j];
  __syncthreads();

  int gid = blockIdx.x * 256 + threadIdx.x;
  if (gid >= NB * HP * WP * 8) return;
  int c0 = (gid & 7) * 8;
  int ppix = gid >> 3;
  int wp = ppix % WP, hp = (ppix / WP) % HP, n = ppix / (WP * HP);
  int h = hp - 2, w = wp - 2;
  _Float16* o = pad + (size_t)ppix * 64 + c0;

  bool oob = (h < 0) | (h >= H) | (w < 0) | (w >= W);
  if (MODE != MODE_REP && oob) {
    h8 z;
#pragma unroll
    for (int j = 0; j < 8; ++j) z[j] = (_Float16)0.f;
    *(h8*)o = z;
    return;
  }
  int hc = min(max(h, 0), H - 1), wc = min(max(w, 0), W - 1);
  const float* tp = t + ((size_t)(n * H + hc) * W + wc) * 64 + c0;
  float xv[8], acc[8];
#pragma unroll
  for (int j = 0; j < 8; ++j) { xv[j] = tp[j]; acc[j] = 0.f; }
  const float* wrow = &sw[c0 * P];
  for (int p = 0; p < P; ++p) {
    float mu = -310.f + 10.f * (float)p;
#pragma unroll
    for (int j = 0; j < 8; ++j) {
      float d = xv[j] - mu;
      float e = __expf(d * d * (-0.005f));          // exp(-d^2/200)
      float term = (MODE == MODE_GRAD) ? e * d : e;
      acc[j] = fmaf(wrow[j * P + p], term, acc[j]);
    }
  }
  h8 rv;
  if (MODE == MODE_GRAD) {
    const float* gp = g + ((size_t)(n * H + h) * W + w) * 64 + c0;
#pragma unroll
    for (int j = 0; j < 8; ++j) rv[j] = (_Float16)(acc[j] * (-0.01f) * gp[j]);
  } else {
#pragma unroll
    for (int j = 0; j < 8; ++j) rv[j] = (_Float16)acc[j];
  }
  *(h8*)o = rv;
}

// ---------------------------------------------------------------------------
// 64->64 5x5 conv as implicit GEMM on WMMA (forward and transpose share this;
// only the weight table differs).  One wave owns a 16co x 32pix strip as two
// 16x16 tiles with independent accumulators; the A fragment (weights) is
// loaded once per tap/chunk and reused by both WMMAs.  Block = 8 waves =
// 64co x 64pix.  K loop = 25 taps x 2 ci-chunks of 32 -> 100 WMMAs / wave.
//   A(co,ci)  : Wt[tap][co][ci]   f16, ci contiguous -> 2x b128 loads
//   B(ci,pix) : pad NHWC          f16, ci contiguous -> 2x b128 loads / tile
// ---------------------------------------------------------------------------
__global__ __launch_bounds__(256) void conv_wmma_kernel(
    const _Float16* __restrict__ pad, const _Float16* __restrict__ Wt,
    const float* __restrict__ bias, float* __restrict__ out) {
  int bid = blockIdx.x;                 // 4 * 128 * 2 = 1024 blocks
  int n = bid >> 8;
  int rem = bid & 255;
  int h = rem >> 1;
  int w0 = (rem & 1) * 64;

  int wv = threadIdx.x >> 5;
  int lane = threadIdx.x & 31;
  int mtile = (wv & 3) * 16;            // co base
  int nh = (wv >> 2) * 32;              // pixel strip base within 64-wide block
  int m = lane & 15;
  int hiOff8 = (lane & 16) ? 8 : 0;     // A K-offset / C-D row offset
  int hiOff16 = (lane & 16) ? 8 : 0;    // B K-offset
  int pw = w0 + nh + m;                 // pixel column of first tile

  v8f c0;
  if (bias) {
    f32x4 b0 = *(const f32x4*)(bias + mtile + hiOff8);
    f32x4 b1 = *(const f32x4*)(bias + mtile + hiOff8 + 4);
#pragma unroll
    for (int r = 0; r < 4; ++r) { c0[r] = b0[r]; c0[r + 4] = b1[r]; }
  } else {
#pragma unroll
    for (int r = 0; r < 8; ++r) c0[r] = 0.f;
  }
  v8f c1 = c0;

#pragma unroll
  for (int kh = 0; kh < 5; ++kh) {
    const _Float16* prow = pad + (size_t)(n * HP + h + kh) * WP * 64;
#pragma unroll
    for (int kw = 0; kw < 5; ++kw) {
      const _Float16* Bp0 = prow + (size_t)(pw + kw) * 64 + hiOff16;
      const _Float16* Bp1 = Bp0 + 16 * 64;
      const _Float16* Ap = Wt + ((size_t)((kh * 5 + kw) * 64) + mtile + m) * 64 + hiOff8;
#pragma unroll
      for (int ch = 0; ch < 2; ++ch) {
        int cb = ch * 32;
        h8 a0 = *(const h8*)(Ap + cb);         // K = cb+kA+0..7
        h8 a1 = *(const h8*)(Ap + cb + 16);    // K = cb+kA+16..23
        h8 p0 = *(const h8*)(Bp0 + cb);        // K = cb+kB+0..7
        h8 p1 = *(const h8*)(Bp0 + cb + 16);   // K = cb+kB+16..23
        h8 q0 = *(const h8*)(Bp1 + cb);
        h8 q1 = *(const h8*)(Bp1 + cb + 16);
        v16h a = __builtin_shufflevector(a0, a1, 0, 1, 2, 3, 4, 5, 6, 7,
                                         8, 9, 10, 11, 12, 13, 14, 15);
        v16h b0v = __builtin_shufflevector(p0, p1, 0, 1, 2, 3, 4, 5, 6, 7,
                                           8, 9, 10, 11, 12, 13, 14, 15);
        v16h b1v = __builtin_shufflevector(q0, q1, 0, 1, 2, 3, 4, 5, 6, 7,
                                           8, 9, 10, 11, 12, 13, 14, 15);
        c0 = __builtin_amdgcn_wmma_f32_16x16x32_f16(false, a, false, b0v,
                                                    (short)0, c0, false, false);
        c1 = __builtin_amdgcn_wmma_f32_16x16x32_f16(false, a, false, b1v,
                                                    (short)0, c1, false, false);
      }
    }
  }
  {
    float* op = out + ((size_t)(n * H + h) * W + pw) * 64 + mtile + hiOff8;
    *(f32x4*)op = __builtin_shufflevector(c0, c0, 0, 1, 2, 3);
    *(f32x4*)(op + 4) = __builtin_shufflevector(c0, c0, 4, 5, 6, 7);
  }
  {
    float* op = out + ((size_t)(n * H + h) * W + pw + 16) * 64 + mtile + hiOff8;
    *(f32x4*)op = __builtin_shufflevector(c1, c1, 0, 1, 2, 3);
    *(f32x4*)(op + 4) = __builtin_shufflevector(c1, c1, 4, 5, 6, 7);
  }
}

// ---------------------------------------------------------------------------
// Last transpose conv (64->1) fused with the output combine:
//   back(h,w) = sum_{kh,kw,co} pad[h+kh][w+kw][co] * f0[co][4-kh][4-kw]
//   out = x - back/255 - exp(lam)*(x - y)
// ---------------------------------------------------------------------------
__global__ __launch_bounds__(256) void final_kernel(
    const float* __restrict__ x, const float* __restrict__ y,
    const float* __restrict__ lam, const float* __restrict__ f0,
    const _Float16* __restrict__ pad, float* __restrict__ out) {
  int gid = blockIdx.x * 256 + threadIdx.x;
  if (gid >= NB * H * W) return;
  int w = gid % W, h = (gid / W) % H, n = gid / (W * H);
  float s = 0.f;
  for (int kh = 0; kh < 5; ++kh) {
    for (int kw = 0; kw < 5; ++kw) {
      const _Float16* pp = pad + ((size_t)(n * HP + h + kh) * WP + (w + kw)) * 64;
      int tapf = (4 - kh) * 5 + (4 - kw);
#pragma unroll
      for (int c8 = 0; c8 < 8; ++c8) {
        h8 pv = *(const h8*)(pp + c8 * 8);
#pragma unroll
        for (int j = 0; j < 8; ++j)
          s = fmaf((float)pv[j], f0[(c8 * 8 + j) * 25 + tapf], s);
      }
    }
  }
  float e = __expf(lam[0]);
  out[gid] = x[gid] - s * (1.f / 255.f) - e * (x[gid] - y[gid]);
}

// ---------------------------------------------------------------------------
extern "C" void kernel_launch(void* const* d_in, const int* in_sizes, int n_in,
                              void* d_out, int out_size, void* d_ws, size_t ws_size,
                              hipStream_t stream) {
  const float* x    = (const float*)d_in[0];
  const float* y    = (const float*)d_in[1];
  const float* lam  = (const float*)d_in[2];
  const float* f0   = (const float*)d_in[3];
  const float* fr   = (const float*)d_in[4];
  const float* bias = (const float*)d_in[5];
  const float* aw   = (const float*)d_in[6];
  float* out = (float*)d_out;

  char* ws = (char*)d_ws;
  size_t off = 0;
  float* t[4];
  for (int i = 0; i < 4; ++i) { t[i] = (float*)(ws + off); off += (size_t)NB * H * W * 64 * 4; }
  _Float16* pad = (_Float16*)(ws + off); off += (size_t)NB * HP * WP * 64 * 2;
  float* g = (float*)(ws + off); off += (size_t)NB * H * W * 64 * 4;
  _Float16* Wf = (_Float16*)(ws + off); off += (size_t)3 * 25 * 64 * 64 * 2;
  _Float16* Wb = (_Float16*)(ws + off); off += (size_t)3 * 25 * 64 * 64 * 2;
  (void)ws_size; (void)in_sizes; (void)n_in; (void)out_size;

  dim3 blk(256);
  const int WS = 25 * 64 * 64;
  int padGrid = (NB * HP * WP * 8 + 255) / 256;
  const int convGrid = NB * H * 2;   // 1024

  prep_weights_kernel<<<(2 * 3 * WS + 255) / 256, blk, 0, stream>>>(fr, Wf, Wb);
  conv0_kernel<<<(NB * H * W * 8 + 255) / 256, blk, 0, stream>>>(x, f0, bias, t[0]);

  // forward diffusion
  for (int i = 0; i < 3; ++i) {
    act_pad_kernel<MODE_REP><<<padGrid, blk, 0, stream>>>(t[i], aw + i * C * P, nullptr, pad);
    conv_wmma_kernel<<<convGrid, blk, 0, stream>>>(pad, Wf + i * WS, bias + (i + 1) * 64, t[i + 1]);
  }
  // backward diffusion
  act_pad_kernel<MODE_ZERO><<<padGrid, blk, 0, stream>>>(t[3], aw + 3 * C * P, nullptr, pad);
  conv_wmma_kernel<<<convGrid, blk, 0, stream>>>(pad, Wb + 2 * WS, nullptr, g);
  for (int i = 2; i >= 1; --i) {
    act_pad_kernel<MODE_GRAD><<<padGrid, blk, 0, stream>>>(t[i], aw + i * C * P, g, pad);
    conv_wmma_kernel<<<convGrid, blk, 0, stream>>>(pad, Wb + (i - 1) * WS, nullptr, g);
  }
  act_pad_kernel<MODE_GRAD><<<padGrid, blk, 0, stream>>>(t[0], aw, g, pad);
  final_kernel<<<(NB * H * W + 255) / 256, blk, 0, stream>>>(x, y, lam, f0, pad, out);
}
